// Attention_87840671138123
// MI455X (gfx1250) — matrix operand order repaired and checked by reading, not compiled, not run on hardware
//
#include <hip/hip_runtime.h>
#include <hip/hip_bf16.h>

// ---------------------------------------------------------------------------
// CDNA5 (gfx1250, wave32) fused causal MHA forward.
//   1) Q/K/V projections : bf16 WMMA GEMM, f32 inputs converted in staging
//                          (Q pre-scaled by 1/sqrt(D); V stored transposed)
//   2) flash attention   : bf16 WMMA for QK^T and P*V, f32 online softmax,
//                          K/V tiles staged with global_load_async_to_lds_b128
//                          directly into WMMA fragment layout (ASYNCcnt)
//   3) output projection : bf16 WMMA GEMM -> f32 out, async A-tile staging
// ---------------------------------------------------------------------------

typedef __attribute__((ext_vector_type(16))) __bf16 v16bf;
typedef __attribute__((ext_vector_type(8)))  float  v8f;

#define B_DIM 2
#define S_DIM 2048
#define D_EMB 1024
#define H_DIM 16
#define E_DIM 64
#define ROWS  (B_DIM * S_DIM)   // 4096

__device__ __forceinline__ v8f wmma_bf16(v16bf a, v16bf b, v8f c) {
  return __builtin_amdgcn_wmma_f32_16x16x32_bf16(
      /*neg_a=*/false, a, /*neg_b=*/false, b,
      /*c_mod=*/(short)0, c, /*reuse_a=*/false, /*reuse_b=*/false);
}

// A-fragment swizzle (16x32 bf16, ISA "16-bit A-Matrix 16x32" table):
//   lane = (m&15) + 16*((k>>3)&1),   elem = 8*(k>>4) + (k&7)
__device__ __forceinline__ int afrag_addr(int m, int k) {
  return ((m & 15) + (((k >> 3) & 1) << 4)) * 16 + (((k >> 4) << 3) + (k & 7));
}
// B-fragment swizzle (32x16 bf16, same k order as the A table):
//   lane = (n&15) + 16*((k>>3)&1),   elem = 8*(k>>4) + (k&7)
__device__ __forceinline__ int bfrag_addr(int k, int n) {
  return ((n & 15) + (((k >> 3) & 1) << 4)) * 16 + (((k >> 4) << 3) + (k & 7));
}

// Async global -> LDS copy of 16 bytes, per-lane LDS destination (ASYNCcnt).
__device__ __forceinline__ void async_copy_b128(uint32_t lds_byte_addr,
                                                const void* gptr) {
  asm volatile("global_load_async_to_lds_b128 %0, %1, off"
               :: "v"(lds_byte_addr), "v"(gptr)
               : "memory");
}
__device__ __forceinline__ void wait_async0() {
  asm volatile("s_wait_asynccnt 0x0" ::: "memory");
}
__device__ __forceinline__ uint32_t lds_addr_of(const void* p) {
  return (uint32_t)(uintptr_t)p;  // LDS aperture: the low 32 bits are the LDS address
}

// ---------------------------------------------------------------------------
// Generic 64x64-tile GEMM, K = 1024, Kstep = 32.
// MODE 0: C[m, h*64+e] = sum_d A_f32[m,d]*W[h,d,e] -> bf16 [B,H,S,E] *cscale
// MODE 1: same math, store transposed            -> bf16 [B,H,E,S] *cscale
// MODE 3: C[m, n] = sum_d A_bf16[m,d]*Wo[n,d]    -> f32  [m,n]
// ---------------------------------------------------------------------------
template <int MODE>
__global__ __launch_bounds__(256) void gemm64_kernel(
    const void* __restrict__ Aptr, const float* __restrict__ Wptr,
    void* __restrict__ Cptr, float cscale) {
  __shared__ alignas(32) __bf16 ldsA[4 * 512];  // 4 row-subtiles of 16x32
  __shared__ alignas(32) __bf16 ldsB[4 * 512];  // 4 col-subtiles of 32x16

  const int tid  = threadIdx.x;
  const int lane = tid & 31;
  const int wave = tid >> 5;
  const int wy   = wave >> 1;      // 0..3 row subtile
  const int wx   = wave & 1;       // 0..1 col pair
  const int mBase = blockIdx.y * 64;
  const int nBase = blockIdx.x * 64;

  v8f acc0{}, acc1{};

  for (int kt = 0; kt < 32; ++kt) {
    const int kBase = kt * 32;
    __syncthreads();

    // ---- stage A tile 64x32 into fragment layout ----
    if constexpr (MODE == 3) {
      // A already bf16: async b128 copies straight into fragment layout.
      // 256 copies of 16B; copy c: subtile, frag-lane r, 8-elem run half.
      const int c    = tid;
      const int sub  = c >> 6;
      const int r    = (c >> 1) & 31;
      const int half = c & 1;
      const int ml   = sub * 16 + (r & 15);
      const int k0   = half * 16 + (r >> 4) * 8;
      async_copy_b128(
          lds_addr_of(&ldsA[0]) + (uint32_t)(sub * 1024 + r * 32 + half * 16),
          (const __bf16*)Aptr + (size_t)(mBase + ml) * D_EMB + kBase + k0);
    } else {
      for (int i = tid; i < 2048; i += 256) {
        const int rl = i >> 5, kl = i & 31;
        const float av =
            ((const float*)Aptr)[(size_t)(mBase + rl) * D_EMB + kBase + kl];
        ldsA[((rl >> 4) << 9) + afrag_addr(rl, kl)] = (__bf16)av;
      }
    }
    // ---- stage B tile 32x64 into fragment layout (f32 -> bf16) ----
    for (int i = tid; i < 2048; i += 256) {
      const int kl = i >> 6, cl = i & 63;
      const int k = kBase + kl, n = nBase + cl;
      float bv;
      if constexpr (MODE == 3)
        bv = Wptr[(size_t)n * D_EMB + k];                       // W_o[n, k]
      else
        bv = Wptr[(size_t)(n >> 6) * (D_EMB * E_DIM) +          // W[h, d, e]
                  (size_t)k * E_DIM + (n & 63)];
      ldsB[((cl >> 4) << 9) + bfrag_addr(kl, cl)] = (__bf16)bv;
    }

    if constexpr (MODE != 3) {
      // prefetch next A tile toward L2 (global_prefetch_b8)
      if (kt + 1 < 32)
        __builtin_prefetch((const char*)Aptr +
                               ((size_t)(mBase + (tid >> 3)) * D_EMB + kBase + 32) * 4,
                           0, 1);
    } else {
      wait_async0();
    }
    __syncthreads();

    const v16bf af = *(const v16bf*)&ldsA[(wy << 9) + lane * 16];
    const v16bf b0 = *(const v16bf*)&ldsB[((wx * 2 + 0) << 9) + lane * 16];
    const v16bf b1 = *(const v16bf*)&ldsB[((wx * 2 + 1) << 9) + lane * 16];
    acc0 = wmma_bf16(af, b0, acc0);
    acc1 = wmma_bf16(af, b1, acc1);
  }

  // ---- store C (C/D layout: row = r + 8*(lane>>4), col = lane&15) ----
#pragma unroll
  for (int cc = 0; cc < 2; ++cc) {
    const v8f& a = cc ? acc1 : acc0;
    const int n = nBase + (wx * 2 + cc) * 16 + (lane & 15);
#pragma unroll
    for (int r = 0; r < 8; ++r) {
      const int m = mBase + wy * 16 + r + ((lane >> 4) << 3);
      if constexpr (MODE == 3) {
        ((float*)Cptr)[(size_t)m * D_EMB + n] = a[r];
      } else {
        const int b = m >> 11, s = m & (S_DIM - 1);
        const int h = n >> 6, e = n & 63;
        const __bf16 val = (__bf16)(a[r] * cscale);
        if constexpr (MODE == 0)
          ((__bf16*)Cptr)[(((size_t)b * H_DIM + h) * S_DIM + s) * E_DIM + e] = val;
        else  // MODE 1: V transposed [B,H,E,S]
          ((__bf16*)Cptr)[(((size_t)b * H_DIM + h) * E_DIM + e) * S_DIM + s] = val;
      }
    }
  }
}

// ---------------------------------------------------------------------------
// Flash attention: grid (S/64, B*H), 128 threads (4 waves, 16 rows each).
// 32-key tiles; causal mask; online softmax in f32; WMMA bf16 matmuls.
// K: [B,H,S,E] bf16 (Q pre-scaled by 1/sqrt(D));  Vt: [B,H,E,S] bf16.
// K/V tiles staged via async global->LDS b128 into fragment layout.
// ---------------------------------------------------------------------------
__global__ __launch_bounds__(128) void flash_attn_kernel(
    const __bf16* __restrict__ Q, const __bf16* __restrict__ K,
    const __bf16* __restrict__ Vt, __bf16* __restrict__ O) {
  __shared__ alignas(32) __bf16 ldsK[4 * 512];  // K^T frags [t_sub*2+e_half]
  __shared__ alignas(32) __bf16 ldsV[4 * 512];  // V frags   [e-subtile]
  __shared__ alignas(32) __bf16 ldsP[4 * 512];  // per-wave P staging

  const int tid  = threadIdx.x;
  const int lane = tid & 31;
  const int wave = tid >> 5;
  const int bh   = blockIdx.y;
  const int rowTile = blockIdx.x * 64;
  const int waveRow = rowTile + wave * 16;
  const size_t headBase  = (size_t)bh * S_DIM * E_DIM;   // K base (S-major)
  const size_t headBaseT = (size_t)bh * E_DIM * S_DIM;   // Vt base (E-major)

  // ---- Q fragments for this wave's 16 rows (A layout, e = 0..63) ----
  v16bf qf0, qf1;
  {
    const int m = waveRow + (lane & 15);
    const int halfb = (lane >> 4) & 1;
    const __bf16* qrow = Q + headBase + (size_t)m * E_DIM;
#pragma unroll
    for (int jj = 0; jj < 16; ++jj) {
      const int k = (halfb << 4) + jj;
      qf0[jj] = qrow[k];
      qf1[jj] = qrow[32 + k];
    }
  }

  v8f o0{}, o1{}, o2{}, o3{};
  float mrow[8], lrow[8];
#pragma unroll
  for (int r = 0; r < 8; ++r) { mrow[r] = -1e30f; lrow[r] = 0.0f; }

  const uint32_t ldsKbase = lds_addr_of(&ldsK[0]);
  const uint32_t ldsVbase = lds_addr_of(&ldsV[0]);
  const int jmax = (rowTile >> 5) + 2;

  for (int j = 0; j < jmax; ++j) {
    const int tBase = j * 32;
    __syncthreads();

    // ---- async-stage K^T fragments: 256 b128 copies (2 per thread) ----
    // frag f = t_sub*2 + e_half32; lane-row r holds t=(r&15)+16*t_sub,
    // elems = 16 consecutive e  -> contiguous in K[t][e].
#pragma unroll
    for (int cc = 0; cc < 2; ++cc) {
      const int c    = tid * 2 + cc;
      const int f    = c >> 6;
      const int r    = (c >> 1) & 31;
      const int half = c & 1;
      const int tl   = (r & 15) + (f >> 1) * 16;
      const int e0   = (f & 1) * 32 + (r >> 4) * 16 + half * 8;
      async_copy_b128(ldsKbase + (uint32_t)(f * 1024 + r * 32 + half * 16),
                      K + headBase + (size_t)(tBase + tl) * E_DIM + e0);
    }
    // ---- async-stage V fragments from transposed V: lane-row r holds
    // e=(r&15)+16*cn, elems = t in A-table order (8-runs contiguous in Vt[e][t]).
#pragma unroll
    for (int cc = 0; cc < 2; ++cc) {
      const int c    = tid * 2 + cc + 256;
      const int cn   = (c >> 6) & 3;
      const int r    = (c >> 1) & 31;
      const int half = c & 1;
      const int e    = (r & 15) + cn * 16;
      const int t0   = tBase + half * 16 + (r >> 4) * 8;
      async_copy_b128(ldsVbase + (uint32_t)(cn * 1024 + r * 32 + half * 16),
                      Vt + headBaseT + (size_t)e * S_DIM + t0);
    }
    wait_async0();
    __syncthreads();

    // ---- scores S = Q K^T for 16 rows x 32 cols (Q pre-scaled) ----
    v8f s0{}, s1{};
    {
      v16bf kb;
      kb = *(const v16bf*)&ldsK[(0 << 9) + lane * 16]; s0 = wmma_bf16(qf0, kb, s0);
      kb = *(const v16bf*)&ldsK[(1 << 9) + lane * 16]; s0 = wmma_bf16(qf1, kb, s0);
      kb = *(const v16bf*)&ldsK[(2 << 9) + lane * 16]; s1 = wmma_bf16(qf0, kb, s1);
      kb = *(const v16bf*)&ldsK[(3 << 9) + lane * 16]; s1 = wmma_bf16(qf1, kb, s1);
    }

    // ---- causal mask + online softmax (half-wave row reductions) ----
    const int colLow = tBase + (lane & 15);
    const int hi8 = (lane >> 4) << 3;
#pragma unroll
    for (int r = 0; r < 8; ++r) {
      const int row = waveRow + r + hi8;
      float v0 = s0[r];
      float v1 = s1[r];
      if (colLow > row)      v0 = -1e30f;
      if (colLow + 16 > row) v1 = -1e30f;

      float tm = fmaxf(v0, v1);
#pragma unroll
      for (int off = 8; off > 0; off >>= 1) tm = fmaxf(tm, __shfl_xor(tm, off, 16));
      const float mn = fmaxf(mrow[r], tm);
      const float sc = __expf(mrow[r] - mn);
      const float p0 = __expf(v0 - mn);
      const float p1 = __expf(v1 - mn);
      float ps = p0 + p1;
#pragma unroll
      for (int off = 8; off > 0; off >>= 1) ps += __shfl_xor(ps, off, 16);
      lrow[r] = lrow[r] * sc + ps;
      mrow[r] = mn;
      o0[r] *= sc; o1[r] *= sc; o2[r] *= sc; o3[r] *= sc;

      // scatter P into A-fragment layout (same-wave LDS, kept in order)
      const int m = r + hi8;
      const int c = lane & 15;
      ldsP[(wave << 9) + afrag_addr(m, c)]      = (__bf16)p0;
      ldsP[(wave << 9) + afrag_addr(m, 16 + c)] = (__bf16)p1;
    }

    // ---- O += P * V ----
    const v16bf pf = *(const v16bf*)&ldsP[(wave << 9) + lane * 16];
    v16bf vb;
    vb = *(const v16bf*)&ldsV[(0 << 9) + lane * 16]; o0 = wmma_bf16(pf, vb, o0);
    vb = *(const v16bf*)&ldsV[(1 << 9) + lane * 16]; o1 = wmma_bf16(pf, vb, o1);
    vb = *(const v16bf*)&ldsV[(2 << 9) + lane * 16]; o2 = wmma_bf16(pf, vb, o2);
    vb = *(const v16bf*)&ldsV[(3 << 9) + lane * 16]; o3 = wmma_bf16(pf, vb, o3);
  }

  // ---- normalize + store attn in [B, S, H*E] bf16 for the output GEMM ----
  const int b = bh >> 4, h = bh & 15;
#pragma unroll
  for (int r = 0; r < 8; ++r) {
    const float inv = 1.0f / lrow[r];
    const int srow = waveRow + r + ((lane >> 4) << 3);
    const size_t base = ((size_t)b * S_DIM + srow) * D_EMB + (size_t)h * E_DIM;
    const int c = lane & 15;
    O[base +  0 + c] = (__bf16)(o0[r] * inv);
    O[base + 16 + c] = (__bf16)(o1[r] * inv);
    O[base + 32 + c] = (__bf16)(o2[r] * inv);
    O[base + 48 + c] = (__bf16)(o3[r] * inv);
  }
}

// ---------------------------------------------------------------------------
extern "C" void kernel_launch(void* const* d_in, const int* in_sizes, int n_in,
                              void* d_out, int out_size, void* d_ws,
                              size_t ws_size, hipStream_t stream) {
  (void)in_sizes; (void)n_in; (void)out_size; (void)ws_size;

  const float* q  = (const float*)d_in[0];
  const float* k  = (const float*)d_in[1];
  const float* v  = (const float*)d_in[2];
  const float* Wq = (const float*)d_in[3];
  const float* Wk = (const float*)d_in[4];
  const float* Wv = (const float*)d_in[5];
  const float* Wo = (const float*)d_in[6];

  const size_t elems = (size_t)ROWS * D_EMB;  // 4,194,304
  __bf16* Qbf = (__bf16*)d_ws;                // [B,H,S,E], pre-scaled
  __bf16* Kbf = Qbf + elems;                  // [B,H,S,E]
  __bf16* Vtb = Kbf + elems;                  // [B,H,E,S] (transposed)
  __bf16* Abf = Vtb + elems;                  // [B,S,H*E]

  const float qscale = 0.03125f;  // 1/sqrt(D_EMB)

  const dim3 gemmGrid(D_EMB / 64, ROWS / 64);  // (16, 64)
  gemm64_kernel<0><<<gemmGrid, 256, 0, stream>>>(q, Wq, Qbf, qscale);
  gemm64_kernel<0><<<gemmGrid, 256, 0, stream>>>(k, Wk, Kbf, 1.0f);
  gemm64_kernel<1><<<gemmGrid, 256, 0, stream>>>(v, Wv, Vtb, 1.0f);

  flash_attn_kernel<<<dim3(S_DIM / 64, B_DIM * H_DIM), 128, 0, stream>>>(
      Qbf, Kbf, Vtb, Abf);

  gemm64_kernel<3><<<gemmGrid, 256, 0, stream>>>(Abf, Wo, (float*)d_out, 1.0f);
}
